// FactorizedAttentionBlock_11424613007724
// MI455X (gfx1250) — matrix operand order repaired and checked by reading, not compiled, not run on hardware
//
#include <hip/hip_runtime.h>
#include <math.h>

// ---------------------------------------------------------------------------
// Problem constants (FactorizedAttentionBlock reference)
// ---------------------------------------------------------------------------
#define C_DIM   768
#define NH_     12
#define HD_     64
#define HID_    1152
#define B__     8
#define T__     8
#define HW_     196
#define NTOK    197
#define BT_     64
#define TM_     (B__ * HW_ * T__)     // 12544 temporal tokens
#define SM_     (BT_ * NTOK)          // 12608 spatial tokens

typedef __attribute__((ext_vector_type(16))) _Float16 v16h;
typedef __attribute__((ext_vector_type(8)))  float    v8f;

union FragU { v16h v; uint4 q[2]; };

// ---------------------------------------------------------------------------
// WMMA fragment loads (wave32, v_wmma_f32_16x16x32_f16)
// A: 16x32 f16 tile from row-major [rows, ld].  Per ISA 7.12.2:
//   lanes 0-15  -> row = lane,    K = {0..7, 16..23}
//   lanes 16-31 -> row = lane-16, K = {8..15, 24..31}
// B: 32x16 f16 tile; B[k][n] = W[n][k] with W row-major [N, K]:
//   lanes 0-15  -> col = lane,    K = {0..7, 16..23}
//   lanes 16-31 -> col = lane-16, K = {8..15, 24..31}
// ---------------------------------------------------------------------------
__device__ __forceinline__ v16h load_frag_a(const _Float16* __restrict__ base,
                                            int ld, int lane, int k0) {
  int r  = lane & 15;
  int kh = (lane >> 4) << 3;
  const _Float16* p = base + (size_t)r * ld + k0 + kh;
  FragU f;
  f.q[0] = *(const uint4*)p;          // K = k0+kh .. +7
  f.q[1] = *(const uint4*)(p + 16);   // K = k0+kh+16 .. +23
  return f.v;
}

__device__ __forceinline__ v16h load_frag_b(const _Float16* __restrict__ base,
                                            int ld, int lane, int k0) {
  int r  = lane & 15;
  int kb = (lane >> 4) << 3;
  const _Float16* p = base + (size_t)r * ld + k0 + kb;
  FragU f;
  f.q[0] = *(const uint4*)p;          // K = k0+kb .. +7
  f.q[1] = *(const uint4*)(p + 16);   // K = k0+kb+16 .. +23
  return f.v;
}

// ---------------------------------------------------------------------------
// Generic WMMA GEMM:  out[M,N] = A[M,K] * W[N,K]^T (+ bias/res/gelu epilogues)
// block = 128 threads = 4 waves (2x2), block tile 64(M) x 128(N),
// wave tile 32(M) x 64(N): per k-step 2 A-frags + 4 B-frags -> 8 WMMAs
// (2x the WMMA-per-load ratio of a 32x32 wave tile).
// Epilogues: 0: f32 plain     1: f16 = acc+bias    2: f32 = acc+bias
//            3: f32 = acc+bias+res                 4: f16 = gelu(acc+bias)
// ---------------------------------------------------------------------------
template <int EPI>
__global__ __launch_bounds__(128) void gemm_wmma(
    const _Float16* __restrict__ A, const _Float16* __restrict__ W,
    const float* __restrict__ bias, const float* __restrict__ res,
    float* __restrict__ outF, _Float16* __restrict__ outH,
    int M, int N, int K)
{
  int tid  = threadIdx.x;
  int wave = tid >> 5;                  // 0..3
  int lane = tid & 31;
  int wm   = wave >> 1;                 // 0..1
  int wn   = wave & 1;                  // 0..1
  int m0   = blockIdx.x * 64  + wm * 32;
  int n0   = blockIdx.y * 128 + wn * 64;

  const _Float16* Ar0 = A + (size_t)m0 * K;
  const _Float16* Ar1 = Ar0 + (size_t)16 * K;
  const _Float16* Wr0 = W + (size_t)n0 * K;
  const _Float16* Wr1 = Wr0 + (size_t)16 * K;
  const _Float16* Wr2 = Wr0 + (size_t)32 * K;
  const _Float16* Wr3 = Wr0 + (size_t)48 * K;

  v8f acc[2][4] = {};
  for (int k0 = 0; k0 < K; k0 += 32) {
    v16h a0 = load_frag_a(Ar0, K, lane, k0);
    v16h a1 = load_frag_a(Ar1, K, lane, k0);
    v16h b0 = load_frag_b(Wr0, K, lane, k0);
    v16h b1 = load_frag_b(Wr1, K, lane, k0);
    v16h b2 = load_frag_b(Wr2, K, lane, k0);
    v16h b3 = load_frag_b(Wr3, K, lane, k0);
    acc[0][0] = __builtin_amdgcn_wmma_f32_16x16x32_f16(false, a0, false, b0, (short)0, acc[0][0], false, false);
    acc[0][1] = __builtin_amdgcn_wmma_f32_16x16x32_f16(false, a0, false, b1, (short)0, acc[0][1], false, false);
    acc[0][2] = __builtin_amdgcn_wmma_f32_16x16x32_f16(false, a0, false, b2, (short)0, acc[0][2], false, false);
    acc[0][3] = __builtin_amdgcn_wmma_f32_16x16x32_f16(false, a0, false, b3, (short)0, acc[0][3], false, false);
    acc[1][0] = __builtin_amdgcn_wmma_f32_16x16x32_f16(false, a1, false, b0, (short)0, acc[1][0], false, false);
    acc[1][1] = __builtin_amdgcn_wmma_f32_16x16x32_f16(false, a1, false, b1, (short)0, acc[1][1], false, false);
    acc[1][2] = __builtin_amdgcn_wmma_f32_16x16x32_f16(false, a1, false, b2, (short)0, acc[1][2], false, false);
    acc[1][3] = __builtin_amdgcn_wmma_f32_16x16x32_f16(false, a1, false, b3, (short)0, acc[1][3], false, false);
  }

  // C/D layout: VGPR e -> row m0 + 16*i + (lane>>4)*8 + e, col n0 + 16*j + (lane&15)
  int rb = (lane >> 4) * 8;
  int cc = lane & 15;
  #pragma unroll
  for (int i = 0; i < 2; ++i) {
    #pragma unroll
    for (int j = 0; j < 4; ++j) {
      int col  = n0 + j * 16 + cc;
      float bv = (EPI != 0) ? bias[col] : 0.0f;
      #pragma unroll
      for (int e = 0; e < 8; ++e) {
        int row    = m0 + i * 16 + rb + e;
        size_t off = (size_t)row * N + col;
        float v    = acc[i][j][e] + bv;
        if (EPI == 0 || EPI == 2) {
          outF[off] = v;
        } else if (EPI == 3) {
          outF[off] = v + res[off];
        } else if (EPI == 1) {
          outH[off] = (_Float16)v;
        } else { // 4: exact GELU, matching jax.nn.gelu(approximate=False)
          float g = 0.5f * v * (1.0f + erff(v * 0.70710678118654752f));
          outH[off] = (_Float16)g;
        }
      }
    }
  }
}

// ---------------------------------------------------------------------------
// fp32 -> f16 weight conversion
// ---------------------------------------------------------------------------
__global__ __launch_bounds__(256) void f32_to_f16(const float* __restrict__ in,
                                                  _Float16* __restrict__ out, int n)
{
  int i = blockIdx.x * 256 + threadIdx.x;
  if (i < n) out[i] = (_Float16)in[i];
}

// ---------------------------------------------------------------------------
// LayerNorm -> f16 rows of 768.  mode 0: row r of `in`.
// mode 1: temporal gather, out row r = (bhw*T + t) reads x[(b*T+t)*197 + s+1].
// ---------------------------------------------------------------------------
__global__ __launch_bounds__(256) void ln_to_f16(const float* __restrict__ in,
                                                 const float* __restrict__ g,
                                                 const float* __restrict__ be,
                                                 _Float16* __restrict__ out, int mode)
{
  __shared__ float rs[256];
  __shared__ float rq[256];
  int r = blockIdx.x;
  const float* src;
  if (mode == 1) {
    int bhw = r >> 3, t = r & 7;
    int b_ = bhw / HW_, s = bhw % HW_;
    src = in + (size_t)((b_ * T__ + t) * NTOK + s + 1) * C_DIM;
  } else {
    src = in + (size_t)r * C_DIM;
  }
  int tid = threadIdx.x;
  float x0 = src[tid], x1 = src[tid + 256], x2 = src[tid + 512];
  rs[tid] = x0 + x1 + x2;
  rq[tid] = x0 * x0 + x1 * x1 + x2 * x2;
  __syncthreads();
  for (int st = 128; st; st >>= 1) {
    if (tid < st) { rs[tid] += rs[tid + st]; rq[tid] += rq[tid + st]; }
    __syncthreads();
  }
  float mean = rs[0] * (1.0f / C_DIM);
  float var  = rq[0] * (1.0f / C_DIM) - mean * mean;
  float rstd = rsqrtf(var + 1e-5f);
  _Float16* o = out + (size_t)r * C_DIM;
  o[tid]       = (_Float16)((x0 - mean) * rstd * g[tid]       + be[tid]);
  o[tid + 256] = (_Float16)((x1 - mean) * rstd * g[tid + 256] + be[tid + 256]);
  o[tid + 512] = (_Float16)((x2 - mean) * rstd * g[tid + 512] + be[tid + 512]);
}

// ---------------------------------------------------------------------------
// Temporal attention: seq length T=8, one 64-thread block per (bhw, head).
// qkv rows indexed (bhw*8 + t); layout [., 2304] = [q|k|v] x heads x 64.
// ---------------------------------------------------------------------------
__global__ __launch_bounds__(64) void attn_temporal(const float* __restrict__ qkv,
                                                    _Float16* __restrict__ O)
{
  int h   = blockIdx.x;
  int bhw = blockIdx.y;
  __shared__ float qs[T__][HD_], ks[T__][HD_], vs[T__][HD_], ps[T__][T__];
  int tid = threadIdx.x;
  for (int i = tid; i < T__ * HD_; i += 64) {
    int t = i >> 6, d = i & 63;
    size_t base = (size_t)(bhw * T__ + t) * (3 * C_DIM) + h * HD_ + d;
    qs[t][d] = qkv[base];
    ks[t][d] = qkv[base + C_DIM];
    vs[t][d] = qkv[base + 2 * C_DIM];
  }
  __syncthreads();
  if (tid < T__) {
    float l[T__];
    float mx = -1e30f;
    for (int j = 0; j < T__; ++j) {
      float s = 0.0f;
      #pragma unroll
      for (int d = 0; d < HD_; ++d) s += qs[tid][d] * ks[j][d];
      s *= 0.125f;                  // hd^-0.5 = 1/8
      l[j] = s;
      mx = fmaxf(mx, s);
    }
    float sum = 0.0f;
    for (int j = 0; j < T__; ++j) { l[j] = expf(l[j] - mx); sum += l[j]; }
    float inv = 1.0f / sum;
    for (int j = 0; j < T__; ++j) ps[tid][j] = l[j] * inv;
  }
  __syncthreads();
  int d = tid;
  for (int t = 0; t < T__; ++t) {
    float o = 0.0f;
    #pragma unroll
    for (int j = 0; j < T__; ++j) o += ps[t][j] * vs[j][d];
    O[(size_t)(bhw * T__ + t) * C_DIM + h * HD_ + d] = (_Float16)o;
  }
}

// ---------------------------------------------------------------------------
// Spatial attention: seq 197, one 256-thread block per (head, bt).
// K/V staged in LDS as f16 (~50 KB of the 320 KB WGP LDS); each wave owns
// q-rows; wave32 shuffles for the softmax reductions.
// ---------------------------------------------------------------------------
__global__ __launch_bounds__(256) void attn_spatial(const float* __restrict__ qkv,
                                                    _Float16* __restrict__ O)
{
  int h  = blockIdx.x;
  int bt = blockIdx.y;
  __shared__ _Float16 Ks[NTOK * HD_];
  __shared__ _Float16 Vs[NTOK * HD_];
  __shared__ float    L[8][200];
  int tid = threadIdx.x, wave = tid >> 5, lane = tid & 31;

  for (int i = tid; i < NTOK * HD_; i += 256) {
    int j = i >> 6, d = i & 63;
    size_t base = (size_t)(bt * NTOK + j) * (3 * C_DIM) + h * HD_ + d;
    Ks[i] = (_Float16)qkv[base + C_DIM];
    Vs[i] = (_Float16)qkv[base + 2 * C_DIM];
  }
  __syncthreads();

  for (int row = wave; row < NTOK; row += 8) {
    size_t qbase = (size_t)(bt * NTOK + row) * (3 * C_DIM) + h * HD_;
    float q[HD_];
    #pragma unroll
    for (int d = 0; d < HD_; ++d) q[d] = qkv[qbase + d];

    float mx = -1e30f;
    for (int j = lane; j < NTOK; j += 32) {
      float s = 0.0f;
      #pragma unroll
      for (int d = 0; d < HD_; ++d) s += q[d] * (float)Ks[j * HD_ + d];
      s *= 0.125f;
      L[wave][j] = s;
      mx = fmaxf(mx, s);
    }
    #pragma unroll
    for (int m = 16; m; m >>= 1) mx = fmaxf(mx, __shfl_xor(mx, m, 32));
    float sum = 0.0f;
    for (int j = lane; j < NTOK; j += 32) {
      float e = expf(L[wave][j] - mx);
      L[wave][j] = e;
      sum += e;
    }
    #pragma unroll
    for (int m = 16; m; m >>= 1) sum += __shfl_xor(sum, m, 32);
    float inv = 1.0f / sum;

    float a0 = 0.0f, a1 = 0.0f;
    int d0 = lane, d1 = lane + 32;
    for (int j = 0; j < NTOK; ++j) {
      float p = L[wave][j];
      a0 += p * (float)Vs[j * HD_ + d0];
      a1 += p * (float)Vs[j * HD_ + d1];
    }
    size_t obase = (size_t)(bt * NTOK + row) * C_DIM + h * HD_;
    O[obase + d0] = (_Float16)(a0 * inv);
    O[obase + d1] = (_Float16)(a1 * inv);
  }
}

// ---------------------------------------------------------------------------
// x_work = x; x_work[:,1:,:] += scatter(temporal_fc output)
// xt2 rows indexed ((b*196 + s)*8 + t) -> x row (b*8+t)*197 + s+1
// ---------------------------------------------------------------------------
__global__ __launch_bounds__(256) void build_xwork(const float* __restrict__ x,
                                                   const float* __restrict__ xt2,
                                                   float* __restrict__ xw)
{
  int r  = blockIdx.x;            // 0..12607
  int bt = r / NTOK, n = r % NTOK;
  int tid = threadIdx.x;
  size_t o = (size_t)r * C_DIM;
  if (n == 0) {
    for (int c = tid; c < C_DIM; c += 256) xw[o + c] = x[o + c];
  } else {
    int b_ = bt >> 3, t = bt & 7;
    size_t ts = (size_t)((b_ * HW_ + (n - 1)) * T__ + t) * C_DIM;
    for (int c = tid; c < C_DIM; c += 256) xw[o + c] = x[o + c] + xt2[ts + c];
  }
}

// ---------------------------------------------------------------------------
// Host-side orchestration
// ---------------------------------------------------------------------------
extern "C" void kernel_launch(void* const* d_in, const int* in_sizes, int n_in,
                              void* d_out, int out_size, void* d_ws, size_t ws_size,
                              hipStream_t stream)
{
  const float* x       = (const float*)d_in[0];
  const float* ln1_g   = (const float*)d_in[1];
  const float* ln1_b   = (const float*)d_in[2];
  const float* qkv_w   = (const float*)d_in[3];
  const float* proj_w  = (const float*)d_in[4];
  const float* proj_b  = (const float*)d_in[5];
  const float* tln_g   = (const float*)d_in[6];
  const float* tln_b   = (const float*)d_in[7];
  const float* tqkv_w  = (const float*)d_in[8];
  const float* tproj_w = (const float*)d_in[9];
  const float* tproj_b = (const float*)d_in[10];
  const float* tfc_w   = (const float*)d_in[11];
  const float* tfc_b   = (const float*)d_in[12];
  const float* ln2_g   = (const float*)d_in[13];
  const float* ln2_b   = (const float*)d_in[14];
  const float* fc1_w   = (const float*)d_in[15];
  const float* fc1_b   = (const float*)d_in[16];
  const float* fc2_w   = (const float*)d_in[17];
  const float* fc2_b   = (const float*)d_in[18];
  float* out = (float*)d_out;

  // --- workspace layout (lifetime-aliased) ---
  char* ws = (char*)d_ws;
  size_t off = 0;
  auto alloc = [&](size_t bytes) -> char* {
    char* p = ws + off;
    off = (off + bytes + 255) & ~(size_t)255;
    return p;
  };
  _Float16* Wtqkv = (_Float16*)alloc((size_t)3 * C_DIM * C_DIM * 2);
  _Float16* Wtprj = (_Float16*)alloc((size_t)C_DIM * C_DIM * 2);
  _Float16* Wtfc  = (_Float16*)alloc((size_t)C_DIM * C_DIM * 2);
  _Float16* Wqkv  = (_Float16*)alloc((size_t)3 * C_DIM * C_DIM * 2);
  _Float16* Wprj  = (_Float16*)alloc((size_t)C_DIM * C_DIM * 2);
  _Float16* Wfc1  = (_Float16*)alloc((size_t)HID_ * C_DIM * 2);
  _Float16* Wfc2  = (_Float16*)alloc((size_t)C_DIM * HID_ * 2);
  _Float16* LNb   = (_Float16*)alloc((size_t)SM_ * C_DIM * 2);   // XT_LN / XLN / X2LN
  float*    QKVb  = (float*)   alloc((size_t)SM_ * 3 * C_DIM * 4); // temporal + spatial qkv
  _Float16* Ob    = (_Float16*)alloc((size_t)SM_ * C_DIM * 2);   // Ot / Os
  _Float16* Hb    = (_Float16*)alloc((size_t)SM_ * HID_ * 2);    // XT1 (f16) / H (f16)
  float*    XT2   = (float*)   alloc((size_t)TM_ * C_DIM * 4);
  float*    Xw    = (float*)   alloc((size_t)SM_ * C_DIM * 4);   // x after temporal add
  (void)ws_size; (void)in_sizes; (void)n_in; (void)out_size;

  auto conv = [&](const float* s, _Float16* d, int n) {
    f32_to_f16<<<(n + 255) / 256, 256, 0, stream>>>(s, d, n);
  };
  conv(tqkv_w, Wtqkv, 3 * C_DIM * C_DIM);
  conv(tproj_w, Wtprj, C_DIM * C_DIM);
  conv(tfc_w,  Wtfc,  C_DIM * C_DIM);
  conv(qkv_w,  Wqkv,  3 * C_DIM * C_DIM);
  conv(proj_w, Wprj,  C_DIM * C_DIM);
  conv(fc1_w,  Wfc1,  HID_ * C_DIM);
  conv(fc2_w,  Wfc2,  C_DIM * HID_);

  // ---- temporal branch ----
  ln_to_f16<<<TM_, 256, 0, stream>>>(x, tln_g, tln_b, LNb, 1);
  gemm_wmma<0><<<dim3(TM_ / 64, (3 * C_DIM) / 128), 128, 0, stream>>>(
      LNb, Wtqkv, nullptr, nullptr, QKVb, nullptr, TM_, 3 * C_DIM, C_DIM);
  attn_temporal<<<dim3(NH_, TM_ / T__), 64, 0, stream>>>(QKVb, Ob);
  gemm_wmma<1><<<dim3(TM_ / 64, C_DIM / 128), 128, 0, stream>>>(
      Ob, Wtprj, tproj_b, nullptr, nullptr, Hb, TM_, C_DIM, C_DIM);   // XT1 f16
  gemm_wmma<2><<<dim3(TM_ / 64, C_DIM / 128), 128, 0, stream>>>(
      Hb, Wtfc, tfc_b, nullptr, XT2, nullptr, TM_, C_DIM, C_DIM);     // XT2 f32
  build_xwork<<<SM_, 256, 0, stream>>>(x, XT2, Xw);

  // ---- spatial attention + residual ----
  ln_to_f16<<<SM_, 256, 0, stream>>>(Xw, ln1_g, ln1_b, LNb, 0);
  gemm_wmma<0><<<dim3(SM_ / 64, (3 * C_DIM) / 128), 128, 0, stream>>>(
      LNb, Wqkv, nullptr, nullptr, QKVb, nullptr, SM_, 3 * C_DIM, C_DIM);
  attn_spatial<<<dim3(NH_, BT_), 256, 0, stream>>>(QKVb, Ob);
  gemm_wmma<3><<<dim3(SM_ / 64, C_DIM / 128), 128, 0, stream>>>(
      Ob, Wprj, proj_b, Xw, out, nullptr, SM_, C_DIM, C_DIM);         // X2 -> d_out

  // ---- MLP + residual ----
  ln_to_f16<<<SM_, 256, 0, stream>>>(out, ln2_g, ln2_b, LNb, 0);
  gemm_wmma<4><<<dim3(SM_ / 64, HID_ / 128), 128, 0, stream>>>(
      LNb, Wfc1, fc1_b, nullptr, nullptr, Hb, SM_, HID_, C_DIM);      // H = gelu(...)
  gemm_wmma<3><<<dim3(SM_ / 64, C_DIM / 128), 128, 0, stream>>>(
      Hb, Wfc2, fc2_b, out, out, nullptr, SM_, C_DIM, HID_);          // out = X2 + fc2(H)
}
